// MLAttention_83683142795696
// MI455X (gfx1250) — matrix operand order repaired and checked by reading, not compiled, not run on hardware
//
#include <hip/hip_runtime.h>
#include <math.h>

typedef __attribute__((ext_vector_type(16))) __bf16 v16bf;
typedef __attribute__((ext_vector_type(8)))  float  v8f;

#define L_SEQ  2048
#define NHEAD  16
#define HDIM   128
#define DMODEL 2048
#define RQ     1536
#define RKV    512
#define BATCH  2
#define MROWS  (BATCH * L_SEQ)   // 4096

union Frag  { uint4 u[2]; v16bf v; };
union Pack8 { __bf16 h[8]; uint4 u; };

// ---------------- fp32 -> bf16 conversion (8 elems -> one 16B store) ----------
__global__ __launch_bounds__(256) void k_cvt(const float* __restrict__ in,
                                             __bf16* __restrict__ out, int n8) {
  int i = blockIdx.x * blockDim.x + threadIdx.x;
  if (i < n8) {
    float4 f0 = ((const float4*)in)[2 * i];
    float4 f1 = ((const float4*)in)[2 * i + 1];
    Pack8 p;
    p.h[0] = (__bf16)f0.x; p.h[1] = (__bf16)f0.y;
    p.h[2] = (__bf16)f0.z; p.h[3] = (__bf16)f0.w;
    p.h[4] = (__bf16)f1.x; p.h[5] = (__bf16)f1.y;
    p.h[6] = (__bf16)f1.z; p.h[7] = (__bf16)f1.w;
    ((uint4*)out)[i] = p.u;
  }
}

// k_up/v_up: [H, R, d] fp32 -> [H*d, R] bf16   (out[h*128+dd, r] = in[h, r, dd])
__global__ __launch_bounds__(256) void k_cvt_upT(const float* __restrict__ in,
                                                 __bf16* __restrict__ out) {
  int i = blockIdx.x * blockDim.x + threadIdx.x;   // over H*HDIM*RKV
  int r  = i % RKV;
  int hd = i / RKV;
  int dd = hd % HDIM;
  int h  = hd / HDIM;
  out[i] = (__bf16)in[((size_t)h * RKV + r) * HDIM + dd];
}

// ---------------- tiled bf16 WMMA GEMM: Y[M,N] = A[M,K] * W[N,K]^T -------------
// 256 threads = 8 waves; block tile 128x128; wave tile 64x32 (4x2 WMMA C tiles).
// Double-buffered LDS (K-step 64) filled by global_load_async_to_lds_b128.
// MODE 0: f32 out row-major (ldc). MODE 1: bf16 out row-major (ldc).
// MODE 2: bf16 out batch-transposed: out[(b*DMODEL + n)*L_SEQ + s], m = b*L_SEQ + s,
//         written coalesced through an LDS bounce.
#define KSTEP 64
#define APITCH 72   // 64 + 8 pad (halves); row = 144 B, 16B-aligned

template <int MODE>
__global__ __launch_bounds__(256) void k_gemm_bf16(const __bf16* __restrict__ A,
                                                   const __bf16* __restrict__ W,
                                                   void* __restrict__ out,
                                                   int K, int lda, int ldw, int ldc) {
  __shared__ __bf16 As[2][128 * APITCH];
  __shared__ __bf16 Bs[2][128 * APITCH];
  const int tid  = threadIdx.x;
  const int wid  = tid >> 5;
  const int lane = tid & 31;
  const int half = lane >> 4;
  const int l16  = lane & 15;
  const int m0   = blockIdx.y * 128;
  const int n0   = blockIdx.x * 128;
  const int wrow = wid >> 2;   // 0..1 -> 64 M-rows each
  const int wcol = wid & 3;    // 0..3 -> 32 N-cols each

  v8f acc[4][2];
#pragma unroll
  for (int i = 0; i < 4; ++i)
#pragma unroll
    for (int j = 0; j < 2; ++j) acc[i][j] = {};

  // async-DMA a 128x64 bf16 tile of A and of W into LDS buffer `buf`
  auto issue = [&](int buf, int k0) {
#pragma unroll
    for (int j = 0; j < 4; ++j) {
      int ci  = tid * 4 + j;        // 0..1023 : 128 rows x 8 chunks of 16B
      int row = ci >> 3, sub = ci & 7;
      unsigned long long ga =
          (unsigned long long)(uintptr_t)(A + (size_t)(m0 + row) * lda + k0 + sub * 8);
      unsigned int la =
          (unsigned int)(uintptr_t)(void*)&As[buf][row * APITCH + sub * 8];
      asm volatile("global_load_async_to_lds_b128 %0, %1, off"
                   :: "v"(la), "v"(ga) : "memory");
      unsigned long long gb =
          (unsigned long long)(uintptr_t)(W + (size_t)(n0 + row) * ldw + k0 + sub * 8);
      unsigned int lb =
          (unsigned int)(uintptr_t)(void*)&Bs[buf][row * APITCH + sub * 8];
      asm volatile("global_load_async_to_lds_b128 %0, %1, off"
                   :: "v"(lb), "v"(gb) : "memory");
    }
  };

  const int nk = K / KSTEP;
  issue(0, 0);
  for (int i = 0; i < nk; ++i) {
    const int p = i & 1;
    asm volatile("s_wait_asynccnt 0x0" ::: "memory");
    __syncthreads();                       // buffer p fully in LDS for all waves
    if (i + 1 < nk) issue(p ^ 1, (i + 1) * KSTEP);  // overlap DMA with compute

#pragma unroll
    for (int ks = 0; ks < 2; ++ks) {       // two 16x16x32 sub-steps per stage
      Frag bf[2];
#pragma unroll
      for (int nt = 0; nt < 2; ++nt) {
        int nr = wcol * 32 + nt * 16 + l16;
        bf[nt].u[0] = *(const uint4*)(&Bs[p][nr * APITCH + ks * 32 + half * 8]);
        bf[nt].u[1] = *(const uint4*)(&Bs[p][nr * APITCH + ks * 32 + 16 + half * 8]);
      }
#pragma unroll
      for (int mt = 0; mt < 4; ++mt) {
        Frag af;
        int mr = wrow * 64 + mt * 16 + l16;
        af.u[0] = *(const uint4*)(&As[p][mr * APITCH + ks * 32 + half * 8]);
        af.u[1] = *(const uint4*)(&As[p][mr * APITCH + ks * 32 + 16 + half * 8]);
#pragma unroll
        for (int nt = 0; nt < 2; ++nt) {
          acc[mt][nt] = __builtin_amdgcn_wmma_f32_16x16x32_bf16(
              false, af.v, false, bf[nt].v, (short)0, acc[mt][nt], false, false);
        }
      }
    }
    __syncthreads();                       // all waves done reading buffer p
  }

  if (MODE != 2) {
#pragma unroll
    for (int mt = 0; mt < 4; ++mt)
#pragma unroll
      for (int nt = 0; nt < 2; ++nt)
#pragma unroll
        for (int r = 0; r < 8; ++r) {
          int m = m0 + wrow * 64 + mt * 16 + r + half * 8;
          int n = n0 + wcol * 32 + nt * 16 + l16;
          float vv = acc[mt][nt][r];
          if (MODE == 0) ((float*)out)[(size_t)m * ldc + n] = vv;
          else           ((__bf16*)out)[(size_t)m * ldc + n] = (__bf16)vv;
        }
  } else {
    // transpose 128x128 tile through LDS, then coalesced 16B row writes of V^T
    __bf16* tb = &As[0][0];                // 128 * 136 * 2B = 34.8 KB, fits
#pragma unroll
    for (int mt = 0; mt < 4; ++mt)
#pragma unroll
      for (int nt = 0; nt < 2; ++nt)
#pragma unroll
        for (int r = 0; r < 8; ++r) {
          int ml = wrow * 64 + mt * 16 + r + half * 8;
          int nl = wcol * 32 + nt * 16 + l16;
          tb[nl * 136 + ml] = (__bf16)acc[mt][nt][r];
        }
    __syncthreads();
    const int b  = m0 >> 11;
    const int s0 = m0 & (L_SEQ - 1);
#pragma unroll
    for (int j = 0; j < 8; ++j) {
      int ci  = tid * 8 + j;               // 2048 chunks: 128 rows x 16 x 16B
      int row = ci >> 4, c8 = ci & 15;
      uint4 v = *(const uint4*)(&tb[row * 136 + c8 * 8]);
      *(uint4*)((__bf16*)out +
                ((size_t)(b * DMODEL + n0 + row)) * L_SEQ + s0 + c8 * 8) = v;
    }
  }
}

// ---------------- flash attention (causal + ALiBi), one wave per 16 queries ----
// q,k: [B, L, H, d] bf16 ; vT: [B, H*d, L] bf16 ; out: [B, L, H, d] bf16
__global__ __launch_bounds__(128) void k_attn(const __bf16* __restrict__ q,
                                              const __bf16* __restrict__ kk,
                                              const __bf16* __restrict__ vT,
                                              __bf16* __restrict__ out) {
  __shared__ __bf16 pbuf[4][16 * 40];
  const int tid  = threadIdx.x;
  const int wid  = tid >> 5;
  const int lane = tid & 31;
  const int half = lane >> 4;
  const int l16  = lane & 15;

  const int qblocks = L_SEQ / 64;
  int gid = blockIdx.x;
  int qb  = gid % qblocks;
  int bh  = gid / qblocks;
  int h   = bh % NHEAD;
  int b   = bh / NHEAD;
  int q0  = qb * 64 + wid * 16;

  const float slope = exp2f(-0.5f * (float)(h + 1));
  const float scale = 0.08838834764831845f;  // 1/sqrt(128)
  const float NEG   = -3.0e38f;

  Frag qf[4];
  {
    const __bf16* qrow = q + (((size_t)(b * L_SEQ + q0 + l16)) * NHEAD + h) * HDIM;
#pragma unroll
    for (int f = 0; f < 4; ++f) {
      qf[f].u[0] = *(const uint4*)(qrow + f * 32 + half * 8);
      qf[f].u[1] = *(const uint4*)(qrow + f * 32 + 16 + half * 8);
    }
  }

  v8f o[8];
#pragma unroll
  for (int i = 0; i < 8; ++i) o[i] = {};
  v8f mrow, lrow;
#pragma unroll
  for (int r = 0; r < 8; ++r) { mrow[r] = NEG; lrow[r] = 0.f; }

  const int nkb = (q0 + 47) >> 5;  // 32-wide key blocks up to the diagonal
  for (int kb32 = 0; kb32 < nkb; ++kb32) {
    const int kb = kb32 * 32;

    // S = Q K^T  (two 16x16 C tiles over 32 keys, K-dim = 128 = 4 WMMA steps)
    v8f s[2]; s[0] = {}; s[1] = {};
#pragma unroll
    for (int nt = 0; nt < 2; ++nt) {
      const __bf16* krow =
          kk + (((size_t)(b * L_SEQ + kb + nt * 16 + l16)) * NHEAD + h) * HDIM;
#pragma unroll
      for (int f = 0; f < 4; ++f) {
        Frag kf;
        kf.u[0] = *(const uint4*)(krow + f * 32 + half * 8);
        kf.u[1] = *(const uint4*)(krow + f * 32 + 16 + half * 8);
        s[nt] = __builtin_amdgcn_wmma_f32_16x16x32_bf16(
            false, qf[f].v, false, kf.v, (short)0, s[nt], false, false);
      }
    }

    // scale + ALiBi + causal mask, row max
    v8f rmax;
#pragma unroll
    for (int r = 0; r < 8; ++r) rmax[r] = NEG;
#pragma unroll
    for (int nt = 0; nt < 2; ++nt)
#pragma unroll
      for (int r = 0; r < 8; ++r) {
        int i = q0 + r + half * 8;
        int j = kb + nt * 16 + l16;
        float sv = s[nt][r] * scale - slope * (float)(i - j);
        sv = (j <= i) ? sv : NEG;
        s[nt][r] = sv;
        rmax[r] = fmaxf(rmax[r], sv);
      }
#pragma unroll
    for (int off = 1; off < 16; off <<= 1)
#pragma unroll
      for (int r = 0; r < 8; ++r)
        rmax[r] = fmaxf(rmax[r], __shfl_xor(rmax[r], off, 32));

    v8f mnew, alpha;
#pragma unroll
    for (int r = 0; r < 8; ++r) {
      mnew[r]  = fmaxf(mrow[r], rmax[r]);
      alpha[r] = __expf(mrow[r] - mnew[r]);
      mrow[r]  = mnew[r];
    }

    // P = exp(S - m), row sums
    v8f rsum = {};
#pragma unroll
    for (int nt = 0; nt < 2; ++nt)
#pragma unroll
      for (int r = 0; r < 8; ++r) {
        float p = __expf(s[nt][r] - mnew[r]);
        s[nt][r] = p;
        rsum[r] += p;
      }
#pragma unroll
    for (int off = 1; off < 16; off <<= 1)
#pragma unroll
      for (int r = 0; r < 8; ++r) rsum[r] += __shfl_xor(rsum[r], off, 32);
#pragma unroll
    for (int r = 0; r < 8; ++r) lrow[r] = lrow[r] * alpha[r] + rsum[r];
#pragma unroll
    for (int d8 = 0; d8 < 8; ++d8) o[d8] = o[d8] * alpha;

    // C-layout P -> A-layout bf16 via per-wave LDS bounce
    __bf16* pb = pbuf[wid];
#pragma unroll
    for (int nt = 0; nt < 2; ++nt)
#pragma unroll
      for (int r = 0; r < 8; ++r)
        pb[(r + half * 8) * 40 + nt * 16 + l16] = (__bf16)s[nt][r];
    asm volatile("s_wait_dscnt 0x0" ::: "memory");
    Frag pf;
    pf.u[0] = *(const uint4*)(&pb[l16 * 40 + half * 8]);
    pf.u[1] = *(const uint4*)(&pb[l16 * 40 + 16 + half * 8]);

    // O += P * V   (vT rows are d-dims, contiguous over keys)
    const __bf16* vbase = vT + ((size_t)(b * DMODEL + h * HDIM)) * L_SEQ;
#pragma unroll
    for (int d8 = 0; d8 < 8; ++d8) {
      const __bf16* vrow = vbase + (size_t)(d8 * 16 + l16) * L_SEQ + kb + half * 8;
      Frag vf;
      vf.u[0] = *(const uint4*)(vrow);
      vf.u[1] = *(const uint4*)(vrow + 16);
      o[d8] = __builtin_amdgcn_wmma_f32_16x16x32_bf16(
          false, pf.v, false, vf.v, (short)0, o[d8], false, false);
    }
  }

  // normalize + store bf16 [B, L, H, d]
  v8f linv;
#pragma unroll
  for (int r = 0; r < 8; ++r) linv[r] = 1.0f / lrow[r];
#pragma unroll
  for (int d8 = 0; d8 < 8; ++d8) {
    v8f ov = o[d8] * linv;
#pragma unroll
    for (int r = 0; r < 8; ++r) {
      size_t idx = (((size_t)(b * L_SEQ + q0 + r + half * 8)) * NHEAD + h) * HDIM +
                   d8 * 16 + l16;
      out[idx] = (__bf16)ov[r];
    }
  }
}

// ---------------- host-side orchestration ----------------
extern "C" void kernel_launch(void* const* d_in, const int* in_sizes, int n_in,
                              void* d_out, int out_size, void* d_ws, size_t ws_size,
                              hipStream_t stream) {
  (void)in_sizes; (void)n_in; (void)out_size; (void)ws_size;
  const float* hs   = (const float*)d_in[0];
  const float* wqd  = (const float*)d_in[1];
  const float* wqu  = (const float*)d_in[2];
  const float* wkvd = (const float*)d_in[3];
  const float* kup  = (const float*)d_in[4];
  const float* vup  = (const float*)d_in[5];
  const float* wo   = (const float*)d_in[6];

  char* ws = (char*)d_ws;
  size_t off = 0;
  auto alloc = [&](size_t elems) {
    __bf16* p = (__bf16*)(ws + off);
    off += ((elems * 2 + 255) / 256) * 256;
    return p;
  };
  __bf16* hs_b   = alloc((size_t)MROWS * DMODEL);
  __bf16* wqd_b  = alloc((size_t)RQ * DMODEL);
  __bf16* wqu_b  = alloc((size_t)DMODEL * RQ);
  __bf16* wkvd_b = alloc((size_t)RKV * DMODEL);
  __bf16* kupT_b = alloc((size_t)DMODEL * RKV);
  __bf16* vupT_b = alloc((size_t)DMODEL * RKV);
  __bf16* wo_b   = alloc((size_t)DMODEL * DMODEL);
  __bf16* qlat_b = alloc((size_t)MROWS * RQ);
  __bf16* q_b    = alloc((size_t)MROWS * DMODEL);
  __bf16* c_b    = alloc((size_t)MROWS * RKV);
  __bf16* k_b    = alloc((size_t)MROWS * DMODEL);
  __bf16* vT_b   = alloc((size_t)MROWS * DMODEL);
  __bf16* at_b   = alloc((size_t)MROWS * DMODEL);

  auto cvt = [&](const float* in, __bf16* outp, size_t n) {
    int n8 = (int)(n / 8);
    k_cvt<<<(n8 + 255) / 256, 256, 0, stream>>>(in, outp, n8);
  };
  cvt(hs,   hs_b,   (size_t)MROWS * DMODEL);
  cvt(wqd,  wqd_b,  (size_t)RQ * DMODEL);
  cvt(wqu,  wqu_b,  (size_t)DMODEL * RQ);
  cvt(wkvd, wkvd_b, (size_t)RKV * DMODEL);
  cvt(wo,   wo_b,   (size_t)DMODEL * DMODEL);
  k_cvt_upT<<<(NHEAD * HDIM * RKV) / 256, 256, 0, stream>>>(kup, kupT_b);
  k_cvt_upT<<<(NHEAD * HDIM * RKV) / 256, 256, 0, stream>>>(vup, vupT_b);

  dim3 blk(256);
  // q_lat = hs @ Wq_down^T
  k_gemm_bf16<1><<<dim3(RQ / 128, MROWS / 128), blk, 0, stream>>>(
      hs_b, wqd_b, qlat_b, DMODEL, DMODEL, DMODEL, RQ);
  // q = q_lat @ Wq_up^T   ([B,L,H,d])
  k_gemm_bf16<1><<<dim3(DMODEL / 128, MROWS / 128), blk, 0, stream>>>(
      qlat_b, wqu_b, q_b, RQ, RQ, RQ, DMODEL);
  // c = hs @ Wkv_down^T
  k_gemm_bf16<1><<<dim3(RKV / 128, MROWS / 128), blk, 0, stream>>>(
      hs_b, wkvd_b, c_b, DMODEL, DMODEL, DMODEL, RKV);
  // k = c @ k_upT^T  ([B,L,H,d])
  k_gemm_bf16<1><<<dim3(DMODEL / 128, MROWS / 128), blk, 0, stream>>>(
      c_b, kupT_b, k_b, RKV, RKV, RKV, DMODEL);
  // vT = (c @ v_upT^T) stored transposed as [B, H*d, L]
  k_gemm_bf16<2><<<dim3(DMODEL / 128, MROWS / 128), blk, 0, stream>>>(
      c_b, vupT_b, vT_b, RKV, RKV, RKV, 0);
  // attention
  k_attn<<<BATCH * NHEAD * (L_SEQ / 64), 128, 0, stream>>>(q_b, k_b, vT_b, at_b);
  // y = attn @ Wo^T  (fp32 out)
  k_gemm_bf16<0><<<dim3(DMODEL / 128, MROWS / 128), blk, 0, stream>>>(
      at_b, wo_b, d_out, DMODEL, DMODEL, DMODEL, DMODEL);
}
